// StockLSTM_51805895525050
// MI455X (gfx1250) — matrix operand order repaired and checked by reading, not compiled, not run on hardware
//
#include <hip/hip_runtime.h>

#define T_STEPS 256
#define KP1 112                 // padded K for layer-1 (H=100 -> 112, chunks {0,32,64,80})
#define KP2 224                 // padded K for layer-2 ([h1|h2] -> 2*112)
#define NP  112                 // padded N per gate block (4 gates -> 448 rows)
#define MT  32                  // batch rows per workgroup
#define NBLK (2048 / MT)        // 64 workgroups

typedef __attribute__((ext_vector_type(16))) __bf16 bf16x16;
typedef __attribute__((ext_vector_type(8)))  __bf16 bf16x8;
typedef __attribute__((ext_vector_type(8)))  float  f32x8;

__device__ __forceinline__ float sigmoid_f(float x) {
    return 1.0f / (1.0f + __expf(-x));
}
__device__ __forceinline__ float tanh_f(float x) {
    float e = __expf(2.0f * x);
    return (e - 1.0f) / (e + 1.0f);
}

__device__ __forceinline__ bf16x16 cat16(bf16x8 lo, bf16x8 hi) {
    return __builtin_shufflevector(lo, hi, 0,1,2,3,4,5,6,7,8,9,10,11,12,13,14,15);
}

// A fragment (16x32 bf16): lanes 0-15 hold M=lane, K={kb..kb+7, kb+16..kb+23};
// lanes 16-31 hold M=lane-16, K={kb+8..kb+15, kb+24..kb+31}.  Two ds_load_b128.
__device__ __forceinline__ bf16x16 load_a(const __bf16* base, int row, int stride,
                                          int kb, int hiHalf) {
    const __bf16* p = base + row * stride + kb + hiHalf * 8;
    bf16x8 lo = *(const bf16x8*)p;
    bf16x8 hi = *(const bf16x8*)(p + 16);
    return cat16(lo, hi);
}

// B fragment (32x16 bf16): lane holds column n=lane&15, K matching load_a
// (lanes 0-15: K={kb..kb+7, kb+16..kb+23}, lanes 16-31: K={kb+8..kb+15, kb+24..kb+31}). Two ds_load_b128.
// All padding is pre-baked into LDS, so the load is unconditional.
__device__ __forceinline__ bf16x16 load_b(const __bf16* base, int row, int stride,
                                          int kb, int hiHalf) {
    const __bf16* p = base + row * stride + kb + hiHalf * 8;
    bf16x8 lo = *(const bf16x8*)p;
    bf16x8 hi = *(const bf16x8*)(p + 16);
    return cat16(lo, hi);
}

extern "C" __global__ __launch_bounds__(256, 1)
void lstm2_wmma_kernel(const float* __restrict__ x,
                       const float* __restrict__ W_ih1, const float* __restrict__ W_hh1,
                       const float* __restrict__ b_ih1, const float* __restrict__ b_hh1,
                       const float* __restrict__ W_ih2, const float* __restrict__ W_hh2,
                       const float* __restrict__ b_ih2, const float* __restrict__ b_hh2,
                       const float* __restrict__ W_lin, const float* __restrict__ b_lin,
                       float* __restrict__ out)
{
    extern __shared__ char smem[];
    __bf16* B2   = (__bf16*)(smem);                         // [448][KP2] bf16 = 200704 B
    __bf16* B1   = (__bf16*)(smem + 200704);                // [448][KP1] bf16 = 100352 B
    __bf16* hbuf = (__bf16*)(smem + 301056);                // [MT][KP2]  bf16 =  14336 B
    float*  bias1 = (float*)(smem + 301056 + MT * KP2 * 2); // [448] f32
    float*  bias2 = bias1 + 448;                            // [448] f32
    float*  w1x   = bias2 + 448;                            // [448] f32

    const int tid   = threadIdx.x;
    const int bBase = blockIdx.x * MT;

    // ---- one-time staging: fp32 global weights -> bf16 LDS.
    //  * N padded to 112 per gate block (rows with h>=100 are zero)
    //  * K padded to 112 per input block; K in [80,96) pre-halved because the
    //    WMMA chunk bases {0,32,64,80} cover that range twice.
    for (int idx = tid; idx < 448 * KP1; idx += 256) {
        int n = idx / KP1, k = idx - n * KP1;
        int gate = n / NP, h = n - gate * NP;
        float v = (h < 100 && k < 100) ? W_hh1[(gate * 100 + h) * 100 + k] : 0.0f;
        if (k >= 80 && k < 96) v *= 0.5f;
        B1[idx] = (__bf16)v;
    }
    for (int idx = tid; idx < 448 * KP2; idx += 256) {
        int n = idx / KP2, k = idx - n * KP2;
        int gate = n / NP, h = n - gate * NP;
        int kk = (k < KP1) ? k : k - KP1;
        const float* W = (k < KP1) ? W_ih2 : W_hh2;
        float v = (h < 100 && kk < 100) ? W[(gate * 100 + h) * 100 + kk] : 0.0f;
        if (kk >= 80 && kk < 96) v *= 0.5f;
        B2[idx] = (__bf16)v;
    }
    for (int n = tid; n < 448; n += 256) {
        int gate = n / NP, h = n - gate * NP;
        bool ok = h < 100;
        int r = gate * 100 + h;
        bias1[n] = ok ? (b_ih1[r] + b_hh1[r]) : 0.0f;
        bias2[n] = ok ? (b_ih2[r] + b_hh2[r]) : 0.0f;
        w1x[n]   = ok ? W_ih1[r] : 0.0f;     // W_ih1 is [400,1]
    }
    for (int i = tid; i < MT * KP2; i += 256) hbuf[i] = (__bf16)0.0f;
    __syncthreads();

    const int lane   = tid & 31;
    const int wave   = tid >> 5;
    const int mt     = wave >> 2;        // M-tile (0..1) owned by this wave
    const int quad   = wave & 3;         // which pair of h-tiles
    const int laneN  = lane & 15;
    const int hiHalf = lane >> 4;

    const int aRow   = mt * 16 + laneN;          // A-fragment row for this lane
    const int mBaseD = mt * 16 + hiHalf * 8;     // C/D fragment M base for this lane

    const int kb1[4] = {0, 32, 64, 80};
    const int kb2[8] = {0, 32, 64, 80, 112, 144, 176, 192};

    // t-invariant scalars hoisted out of the time loop (LDS -> registers)
    float bs1v[2][4], wx1v[2][4], bs2v[2][4];
    #pragma unroll
    for (int s = 0; s < 2; ++s) {
        const int hT = quad * 2 + s;
        #pragma unroll
        for (int g = 0; g < 4; ++g) {
            if (hT < 7) {
                const int r = g * NP + hT * 16 + laneN;
                bs1v[s][g] = bias1[r];
                wx1v[s][g] = w1x[r];
                bs2v[s][g] = bias2[r];
            } else {
                bs1v[s][g] = 0.0f; wx1v[s][g] = 0.0f; bs2v[s][g] = 0.0f;
            }
        }
    }

    // persistent cell state, fully register-resident (lane-local thanks to the
    // gate-aligned N decomposition)
    float c1s[2][8], c2s[2][8];
    #pragma unroll
    for (int s = 0; s < 2; ++s)
        #pragma unroll
        for (int j = 0; j < 8; ++j) { c1s[s][j] = 0.0f; c2s[s][j] = 0.0f; }

    for (int t = 0; t < T_STEPS; ++t) {
        // x fragment for this lane's 8 output rows (x is [B][T] row-major)
        float xv[8];
        #pragma unroll
        for (int j = 0; j < 8; ++j)
            xv[j] = x[(bBase + mBaseD + j) * T_STEPS + t];
        if (t + 1 < T_STEPS)
            __builtin_prefetch(&x[(bBase + mBaseD) * T_STEPS + t + 1], 0, 1);

        // ================= Layer 1: gates1 = h1_prev @ Wh1^T + x*w1 + b =======
        bf16x16 a1[4];
        #pragma unroll
        for (int c = 0; c < 4; ++c) a1[c] = load_a(hbuf, aRow, KP2, kb1[c], hiHalf);

        float h1n[2][8];
        #pragma unroll
        for (int s = 0; s < 2; ++s) {
            const int hT = quad * 2 + s;             // 0..7 (7 = dummy slot)
            if (hT < 7) {                            // wave-uniform skip of dummy
                const int nrow = hT * 16 + laneN;    // h index within gate block

                f32x8 acc[4];
                #pragma unroll
                for (int g = 0; g < 4; ++g)
                    #pragma unroll
                    for (int j = 0; j < 8; ++j)
                        acc[g][j] = bs1v[s][g] + xv[j] * wx1v[s][g];

                #pragma unroll
                for (int c = 0; c < 4; ++c) {
                    #pragma unroll
                    for (int g = 0; g < 4; ++g) {
                        bf16x16 b = load_b(B1, g * NP + nrow, KP1, kb1[c], hiHalf);
                        acc[g] = __builtin_amdgcn_wmma_f32_16x16x32_bf16(
                            false, a1[c], false, b, (short)0, acc[g], false, false);
                    }
                }
                #pragma unroll
                for (int j = 0; j < 8; ++j) {
                    float ig = sigmoid_f(acc[0][j]);
                    float fg = sigmoid_f(acc[1][j]);
                    float gg = tanh_f(acc[2][j]);
                    float og = sigmoid_f(acc[3][j]);
                    float cc = fg * c1s[s][j] + ig * gg;
                    c1s[s][j] = cc;
                    h1n[s][j] = og * tanh_f(cc);
                }
            }
        }
        __syncthreads();                 // all waves done reading h1_prev
        #pragma unroll
        for (int s = 0; s < 2; ++s) {
            const int hT = quad * 2 + s;
            if (hT < 7) {
                const int hcol = hT * 16 + laneN;
                #pragma unroll
                for (int j = 0; j < 8; ++j)
                    hbuf[(mBaseD + j) * KP2 + hcol] = (__bf16)h1n[s][j];
            }
        }
        __syncthreads();                 // h1_new visible

        // ====== Layer 2: gates2 = [h1_new|h2_prev] @ [Wi2^T;Wh2^T] + b ========
        bf16x16 a2[8];
        #pragma unroll
        for (int c = 0; c < 8; ++c) a2[c] = load_a(hbuf, aRow, KP2, kb2[c], hiHalf);

        float h2n[2][8];
        #pragma unroll
        for (int s = 0; s < 2; ++s) {
            const int hT = quad * 2 + s;
            if (hT < 7) {
                const int nrow = hT * 16 + laneN;

                f32x8 acc[4];
                #pragma unroll
                for (int g = 0; g < 4; ++g)
                    #pragma unroll
                    for (int j = 0; j < 8; ++j)
                        acc[g][j] = bs2v[s][g];

                #pragma unroll
                for (int c = 0; c < 8; ++c) {
                    #pragma unroll
                    for (int g = 0; g < 4; ++g) {
                        bf16x16 b = load_b(B2, g * NP + nrow, KP2, kb2[c], hiHalf);
                        acc[g] = __builtin_amdgcn_wmma_f32_16x16x32_bf16(
                            false, a2[c], false, b, (short)0, acc[g], false, false);
                    }
                }
                #pragma unroll
                for (int j = 0; j < 8; ++j) {
                    float ig = sigmoid_f(acc[0][j]);
                    float fg = sigmoid_f(acc[1][j]);
                    float gg = tanh_f(acc[2][j]);
                    float og = sigmoid_f(acc[3][j]);
                    float cc = fg * c2s[s][j] + ig * gg;
                    c2s[s][j] = cc;
                    h2n[s][j] = og * tanh_f(cc);
                }
            }
        }
        __syncthreads();                 // all waves done reading h2_prev
        #pragma unroll
        for (int s = 0; s < 2; ++s) {
            const int hT = quad * 2 + s;
            if (hT < 7) {
                const int hcol = hT * 16 + laneN;
                #pragma unroll
                for (int j = 0; j < 8; ++j)
                    hbuf[(mBaseD + j) * KP2 + KP1 + hcol] = (__bf16)h2n[s][j];
            }
        }
        __syncthreads();                 // h2_new visible for next timestep
    }

    // ---- linear head: out[b] = h2 . W_lin + b_lin ----
    if (tid < MT) {
        float s = b_lin[0];
        #pragma unroll 4
        for (int k = 0; k < 100; ++k)
            s += (float)hbuf[tid * KP2 + KP1 + k] * W_lin[k];
        out[bBase + tid] = s;
    }
}

extern "C" void kernel_launch(void* const* d_in, const int* in_sizes, int n_in,
                              void* d_out, int out_size, void* d_ws, size_t ws_size,
                              hipStream_t stream) {
    (void)in_sizes; (void)n_in; (void)d_ws; (void)ws_size; (void)out_size;
    const float* x     = (const float*)d_in[0];
    const float* W_ih1 = (const float*)d_in[1];
    const float* W_hh1 = (const float*)d_in[2];
    const float* b_ih1 = (const float*)d_in[3];
    const float* b_hh1 = (const float*)d_in[4];
    const float* W_ih2 = (const float*)d_in[5];
    const float* W_hh2 = (const float*)d_in[6];
    const float* b_ih2 = (const float*)d_in[7];
    const float* b_hh2 = (const float*)d_in[8];
    const float* W_lin = (const float*)d_in[9];
    const float* b_lin = (const float*)d_in[10];
    float* out = (float*)d_out;

    const size_t shmem = 200704u + 100352u + (size_t)MT * KP2 * 2 + 3u * 448u * 4u; // 320768 B
    (void)hipFuncSetAttribute((const void*)lstm2_wmma_kernel,
                              hipFuncAttributeMaxDynamicSharedMemorySize, (int)shmem);
    lstm2_wmma_kernel<<<NBLK, 256, shmem, stream>>>(
        x, W_ih1, W_hh1, b_ih1, b_hh1, W_ih2, W_hh2, b_ih2, b_hh2, W_lin, b_lin, out);
}
